// RNN_64364379898035
// MI455X (gfx1250) — compile-verified, matrix operand order checked by reading, not run on hardware
//
#include <hip/hip_runtime.h>

// ---------------------------------------------------------------------------
// LSTM char-RNN forward, CDNA5 (gfx1250), wave32 WMMA f16.
// One block owns 16 batch rows + ALL 1024 gate columns -> no inter-block sync
// across the 40 recurrent steps; h/c never leave the WGP.
// Round 5: launder a loop-variant zero INDEX (not the pointer) so the B loads
// keep global address-space provenance (global_load_b128, LOADcnt only)
// instead of flat_load_b128 (LOADcnt+DScnt coupling with the LDS A loads),
// while still defeating LICM hoist+spill. sched_barrier per K-chunk caps
// scheduler look-ahead to one chunk.
// ---------------------------------------------------------------------------

typedef __attribute__((ext_vector_type(16))) _Float16 v16h;
typedef __attribute__((ext_vector_type(8)))  float    v8f;

#define BATCH 8192
#define SEQ   40
#define HID   256
#define NCH   128
#define NG    1024       // 4*HID gate columns
#define MB    16         // batch rows per block
#define HSTR  264        // h_lds row stride in f16 elems (16B-aligned, banks spread)

#if defined(__has_builtin) && __has_builtin(__builtin_amdgcn_sched_barrier)
#define SCHED_FENCE() __builtin_amdgcn_sched_barrier(0)
#else
#define SCHED_FENCE() asm volatile("" ::: "memory")
#endif

union ABFrag {
  v16h v;
  unsigned int u[8];
  uint4 q[2];
};

// Element e (0..15) of a WMMA 16-bit A/B fragment for lane `lane` maps to
// K = k_off(lane, e) within the 32-wide K chunk (ISA 7.12.2 16-bit layout).
__device__ __forceinline__ int k_off(int lane, int e) {
  int v = e >> 1;
  return ((v & 4) << 2)        // VGPR 4..7 -> K+16
       + ((lane >> 4) << 3)    // lanes 16..31 -> K+8
       + ((v & 3) << 1)
       + (e & 1);
}

__device__ __forceinline__ unsigned short f2h(float x) {
  union { _Float16 h; unsigned short u; } c;
  c.h = (_Float16)x;
  return c.u;
}

__device__ __forceinline__ float sigm(float x) {
  return 1.0f / (1.0f + __expf(-x));
}
__device__ __forceinline__ float tanh_f(float x) {
  float t = __expf(2.0f * x);
  return (t - 1.0f) / (t + 1.0f);
}

__device__ __forceinline__ void load_b(ABFrag& b, const uint4* bp) {
  b.q[0] = bp[0];
  b.q[1] = bp[1];
}

__device__ __forceinline__ void load_a(ABFrag& a, const unsigned short* hrow,
                                       int lane, int kc) {
#pragma unroll
  for (int p = 0; p < 8; ++p) {
    int K = kc * 32 + k_off(lane, 2 * p);
    a.u[p] = *(const unsigned int*)(hrow + K);
  }
}

// ---------------------------------------------------------------------------
// Pack kernels: rewrite weights into per-lane WMMA fragment order (f16 pairs
// packed into u32), so a B fragment load is 2 coalesced b128 loads.
// ---------------------------------------------------------------------------

// Wxb[v][n] = Wx[v][n] + b[n]   (one-hot GEMM + bias folded into a gather)
__global__ void pack_wxb(const float* __restrict__ Wx, const float* __restrict__ b,
                         float* __restrict__ Wxb) {
  int i = blockIdx.x * blockDim.x + threadIdx.x;
  if (i < NCH * NG) Wxb[i] = Wx[i] + b[i % NG];
}

// Wh (HID x NG, row-major) -> Whp[nt][kc][lane][8 u32], nt: 64 col-tiles of 16
__global__ void pack_wh(const float* __restrict__ Wh, unsigned int* __restrict__ Whp) {
  int idx = blockIdx.x * blockDim.x + threadIdx.x;   // 0 .. 64*8*32-1
  if (idx >= 64 * 8 * 32) return;
  int lane = idx & 31;
  int kc   = (idx >> 5) & 7;
  int nt   = idx >> 8;
  int n    = nt * 16 + (lane & 15);
#pragma unroll
  for (int p = 0; p < 8; ++p) {
    int K = kc * 32 + k_off(lane, 2 * p);
    unsigned int lo = f2h(Wh[(size_t)K * NG + n]);
    unsigned int hi = f2h(Wh[(size_t)(K + 1) * NG + n]);
    Whp[(size_t)idx * 8 + p] = lo | (hi << 16);
  }
}

// Wd (HID x NCH) -> Wdp[nt][kc][lane][8 u32], nt: 8 col-tiles of 16
__global__ void pack_wd(const float* __restrict__ Wd, unsigned int* __restrict__ Wdp) {
  int idx = blockIdx.x * blockDim.x + threadIdx.x;   // 0 .. 8*8*32-1
  if (idx >= 8 * 8 * 32) return;
  int lane = idx & 31;
  int kc   = (idx >> 5) & 7;
  int nt   = idx >> 8;
  int n    = nt * 16 + (lane & 15);
#pragma unroll
  for (int p = 0; p < 8; ++p) {
    int K = kc * 32 + k_off(lane, 2 * p);
    unsigned int lo = f2h(Wd[(size_t)K * NCH + n]);
    unsigned int hi = f2h(Wd[(size_t)(K + 1) * NCH + n]);
    Wdp[(size_t)idx * 8 + p] = lo | (hi << 16);
  }
}

// ---------------------------------------------------------------------------
// Main persistent-tile LSTM kernel. 512 threads = 16 waves.
//   wave w owns hidden slice [16w, 16w+16) across gates i,f,g,o
//   -> 4 accumulator tiles (v8f = 32 VGPRs); c state in 8 VGPRs/lane.
// ---------------------------------------------------------------------------
__global__ __launch_bounds__(512)
void lstm_kernel(const int* __restrict__ inputs,
                 const float* __restrict__ Wxb,
                 const unsigned int* __restrict__ Whp,
                 const unsigned int* __restrict__ Wdp,
                 const float* __restrict__ bd,
                 float* __restrict__ out) {
  __shared__ unsigned short h_lds[MB * HSTR];     // h as f16, [row][hid]
  __shared__ int   chars_lds[MB * SEQ];
  __shared__ float logit_lds[MB][NCH + 4];

  const int tid  = threadIdx.x;
  const int lane = tid & 31;
  const int wave = tid >> 5;          // 0..15
  const int lM   = lane & 15;
  const int lH   = lane >> 4;         // 0/1
  const int row0 = blockIdx.x * MB;
  const int hid0 = wave * 16;         // this wave's hidden slice

  // Stage this block's characters for all timesteps.
  for (int i = tid; i < MB * SEQ; i += blockDim.x) {
    int m = i / SEQ, t = i - m * SEQ;
    chars_lds[i] = inputs[(size_t)(row0 + m) * SEQ + t];
  }
  // h0 = 0
  for (int i = tid; i < MB * HSTR; i += blockDim.x) h_lds[i] = 0;

  float cst[8];
#pragma unroll
  for (int r = 0; r < 8; ++r) cst[r] = 0.0f;

  // Per-wave B-fragment base pointers: tile nt = g*16 + wave.
  // Per (nt,kc): 32 lanes * 8 u32 = 64 uint4; kc stride folds to imm offset.
  const uint4* bbase[4];
#pragma unroll
  for (int g = 0; g < 4; ++g)
    bbase[g] = (const uint4*)Whp + ((size_t)(g * 16 + wave) * 8 * 32 + lane) * 2;

  const unsigned short* hrow = h_lds + (size_t)lM * HSTR;   // A-row base

  __syncthreads();

  for (int t = 0; t < SEQ; ++t) {
    // Loop-variant zero: opaque to the optimizer, so the Whp fragment loads
    // below cannot be hoisted out of the t-loop by LICM (which previously
    // spilled 2KB of hoisted fragments to scratch). Indexing with the zero
    // (instead of laundering the pointer itself) preserves address-space
    // provenance -> global_load_b128, not flat_load_b128.
    int lz = 0;
    asm volatile("" : "+v"(lz));
    const uint4* bbl[4];
#pragma unroll
    for (int g = 0; g < 4; ++g) bbl[g] = bbase[g] + lz;

    // --- init accumulators with (Wx + b)[char, col]  (one-hot GEMM + bias) ---
    int ch[8];
#pragma unroll
    for (int r = 0; r < 8; ++r) ch[r] = chars_lds[(r + 8 * lH) * SEQ + t];

    v8f acc[4];
#pragma unroll
    for (int g = 0; g < 4; ++g) {
      int col = g * HID + hid0 + lM;
#pragma unroll
      for (int r = 0; r < 8; ++r)
        acc[g][r] = Wxb[(size_t)ch[r] * NG + col];
    }

    // --- z += h @ Wh : K = 256 in 8 chunks of 32, WMMA f32_16x16x32_f16 ---
    // Explicit 2-deep pipeline: prefetch B[kc+1] while WMMA consumes B[kc].
    // SCHED_FENCE caps scheduler look-ahead to one chunk.
    ABFrag bq[2][4];
#pragma unroll
    for (int g = 0; g < 4; ++g) load_b(bq[0][g], bbl[g]);

#pragma unroll
    for (int kc = 0; kc < 8; ++kc) {
      const int cur = kc & 1;
      if (kc < 7) {
#pragma unroll
        for (int g = 0; g < 4; ++g)
          load_b(bq[cur ^ 1][g], bbl[g] + (kc + 1) * 64);
      }
      ABFrag a;
      load_a(a, hrow, lane, kc);
#pragma unroll
      for (int g = 0; g < 4; ++g)
        acc[g] = __builtin_amdgcn_wmma_f32_16x16x32_f16(
            false, a.v, false, bq[cur][g].v, (short)0, acc[g], false, false);
      SCHED_FENCE();
    }

    __syncthreads();   // everyone done reading h_lds before it is rewritten

    // --- gate nonlinearities (keras order i,f,g,o), update c, write h ---
    {
      int hid = hid0 + lM;
#pragma unroll
      for (int r = 0; r < 8; ++r) {
        float zi = acc[0][r];
        float zf = acc[1][r];
        float zg = acc[2][r];
        float zo = acc[3][r];
        float cc = sigm(zf) * cst[r] + sigm(zi) * tanh_f(zg);
        cst[r] = cc;
        float hh = sigm(zo) * tanh_f(cc);
        int M = r + 8 * lH;
        h_lds[(size_t)M * HSTR + hid] = f2h(hh);
      }
    }

    __syncthreads();   // h_lds complete for next step
  }

  // --- logits = h @ Wd + bd : waves 0..7 each compute one 16-col tile ---
  if (wave < 8) {
    v8f lac;
    float bv = bd[wave * 16 + lM];
#pragma unroll
    for (int r = 0; r < 8; ++r) lac[r] = bv;
#pragma unroll
    for (int kc = 0; kc < 8; ++kc) {
      ABFrag a;
      load_a(a, hrow, lane, kc);
      ABFrag b;
      load_b(b, (const uint4*)Wdp + ((size_t)(wave * 8 + kc) * 32 + lane) * 2);
      lac = __builtin_amdgcn_wmma_f32_16x16x32_f16(
          false, a.v, false, b.v, (short)0, lac, false, false);
      SCHED_FENCE();
    }
#pragma unroll
    for (int r = 0; r < 8; ++r)
      logit_lds[r + 8 * lH][wave * 16 + lM] = lac[r];
  }
  __syncthreads();

  // --- softmax: one thread per batch row (tiny: 16 x 128) ---
  if (tid < MB) {
    int m = tid;
    float mx = -1e30f;
    for (int j = 0; j < NCH; ++j) mx = fmaxf(mx, logit_lds[m][j]);
    float sum = 0.0f;
    for (int j = 0; j < NCH; ++j) {
      float e = __expf(logit_lds[m][j] - mx);
      logit_lds[m][j] = e;
      sum += e;
    }
    float inv = 1.0f / sum;
    for (int j = 0; j < NCH; ++j)
      out[(size_t)(row0 + m) * NCH + j] = logit_lds[m][j] * inv;
  }
}

// ---------------------------------------------------------------------------
extern "C" void kernel_launch(void* const* d_in, const int* in_sizes, int n_in,
                              void* d_out, int out_size, void* d_ws, size_t ws_size,
                              hipStream_t stream) {
  const int*   inputs = (const int*)  d_in[0];
  const float* Wx     = (const float*)d_in[1];
  const float* Wh     = (const float*)d_in[2];
  const float* b      = (const float*)d_in[3];
  const float* Wd     = (const float*)d_in[4];
  const float* bd     = (const float*)d_in[5];

  // Workspace layout: Whp (512KB) | Wxb (512KB) | Wdp (64KB)
  unsigned int* Whp = (unsigned int*)d_ws;
  float*        Wxb = (float*)((char*)d_ws + (512u << 10));
  unsigned int* Wdp = (unsigned int*)((char*)d_ws + (1024u << 10));

  pack_wxb<<<(NCH * NG + 255) / 256, 256, 0, stream>>>(Wx, b, Wxb);
  pack_wh <<<(64 * 8 * 32 + 255) / 256, 256, 0, stream>>>(Wh, Whp);
  pack_wd <<<( 8 * 8 * 32 + 255) / 256, 256, 0, stream>>>(Wd, Wdp);

  lstm_kernel<<<BATCH / MB, 512, 0, stream>>>(inputs, Wxb, Whp, Wdp, bd,
                                              (float*)d_out);
}
